// ExtraSampleLayer_57226144252766
// MI455X (gfx1250) — matrix operand order repaired and checked by reading, not compiled, not run on hardware
//
#include <hip/hip_runtime.h>
#include <hip/hip_bf16.h>
#include <math.h>

typedef __attribute__((ext_vector_type(16))) __bf16 v16bf;
typedef __attribute__((ext_vector_type(8)))  float  v8f;

#define TOK   65536       // B*N
#define NTOK  16384       // N
#define DDIM  256
#define CIN   320
#define HID   1024
#define HM    128
#define SRCW  512

__device__ __forceinline__ float wave_sum(float v) {
#pragma unroll
  for (int off = 16; off > 0; off >>= 1) v += __shfl_xor(v, off, 32);
  return v;
}

// ---------------- zero workspace region ----------------
__global__ void zero_kernel(float* __restrict__ p, int n) {
  int i = blockIdx.x * 256 + threadIdx.x;
  if (i < n) p[i] = 0.f;
}

// ---------------- copy x -> out1[:, 0:N, :] ----------------
__global__ void copyx_kernel(const float4* __restrict__ x, float4* __restrict__ out) {
  int i = blockIdx.x * 256 + threadIdx.x;          // < 4,194,304
  int t = i >> 6, d4 = i & 63;
  int b = t >> 14, n = t & 16383;
  out[((size_t)(b * 32768 + n)) * 64 + d4] = x[i];
}

// ---------------- delta = (LN(x)@w_delta + b)*0.01 ; write loc & loc_extra ----------------
__global__ void delta_kernel(const float* __restrict__ x, const float* __restrict__ loc,
                             const float* __restrict__ g, const float* __restrict__ bt,
                             const float* __restrict__ wd, const float* __restrict__ bd,
                             float* __restrict__ out2) {
  int wave = threadIdx.x >> 5, lane = threadIdx.x & 31;
  int t = blockIdx.x * 8 + wave;
  const float* xp = x + (size_t)t * DDIM + lane * 8;
  float v[8], s = 0.f, sq = 0.f;
#pragma unroll
  for (int i = 0; i < 8; ++i) { v[i] = xp[i]; s += v[i]; sq += v[i] * v[i]; }
  s = wave_sum(s); sq = wave_sum(sq);
  float mean = s * (1.f / 256.f);
  float var  = sq * (1.f / 256.f) - mean * mean;
  float rstd = rsqrtf(var + 1e-5f);
  float d0 = 0.f, d1 = 0.f;
#pragma unroll
  for (int i = 0; i < 8; ++i) {
    int k = lane * 8 + i;
    float xn = (v[i] - mean) * rstd * g[k] + bt[k];
    d0 += xn * wd[k * 2 + 0];
    d1 += xn * wd[k * 2 + 1];
  }
  d0 = wave_sum(d0); d1 = wave_sum(d1);
  if (lane == 0) {
    int b = t >> 14, n = t & 16383;
    float l0 = loc[(size_t)t * 2 + 0], l1 = loc[(size_t)t * 2 + 1];
    float e0 = fminf(fmaxf(l0 + (d0 + bd[0]) * 0.01f, 0.f), 1.f);
    float e1 = fminf(fmaxf(l1 + (d1 + bd[1]) * 0.01f, 0.f), 1.f);
    size_t o1 = (size_t)(b * 32768 + n) * 2;
    out2[o1] = l0; out2[o1 + 1] = l1;
    size_t o2 = (size_t)(b * 32768 + 16384 + n) * 2;
    out2[o2] = e0; out2[o2 + 1] = e1;
  }
}

// ---------------- 4x4 patch sample + conv einsum + LN -> act[:,256:320] (bf16) ----------------
__global__ void patch_kernel(const float* __restrict__ src, const float* __restrict__ out2,
                             const float* __restrict__ cw, const float* __restrict__ cb,
                             const float* __restrict__ g2, const float* __restrict__ b2,
                             __bf16* __restrict__ act) {
  __shared__ float smem[8][48];
  int wave = threadIdx.x >> 5, lane = threadIdx.x & 31;
  int t = blockIdx.x * 8 + wave;
  int b = t >> 14, n = t & 16383;
  const float* le = out2 + (size_t)(b * 32768 + 16384 + n) * 2;
  float lex = le[0], ley = le[1];
  for (int s = lane; s < 48; s += 32) {
    int ch = s >> 4, p = s & 15, kh = p >> 2, kw = p & 3;
    float gx = (lex + ((float)kw - 1.5f) * (1.f / 511.f)) * 512.f - 0.5f;
    float gy = (ley + ((float)kh - 1.5f) * (1.f / 511.f)) * 512.f - 0.5f;
    float x0 = floorf(gx), y0 = floorf(gy);
    float wx = gx - x0, wy = gy - y0;
    int xi = (int)x0, yi = (int)y0;
    const float* sp = src + (size_t)(b * 3 + ch) * (SRCW * SRCW);
    float acc = 0.f;
    bool vx0 = (xi >= 0) & (xi < SRCW), vx1 = (xi + 1 >= 0) & (xi + 1 < SRCW);
    bool vy0 = (yi >= 0) & (yi < SRCW), vy1 = (yi + 1 >= 0) & (yi + 1 < SRCW);
    if (vx0 && vy0) acc += sp[yi * SRCW + xi] * (1.f - wx) * (1.f - wy);
    if (vx1 && vy0) acc += sp[yi * SRCW + xi + 1] * wx * (1.f - wy);
    if (vx0 && vy1) acc += sp[(yi + 1) * SRCW + xi] * (1.f - wx) * wy;
    if (vx1 && vy1) acc += sp[(yi + 1) * SRCW + xi + 1] * wx * wy;
    smem[wave][s] = acc;
  }
  __syncthreads();
  float a0 = cb[lane], a1 = cb[lane + 32];
#pragma unroll
  for (int s = 0; s < 48; ++s) {
    float pv = smem[wave][s];
    a0 += pv * cw[lane * 48 + s];
    a1 += pv * cw[(lane + 32) * 48 + s];
  }
  float s  = wave_sum(a0 + a1);
  float sq = wave_sum(a0 * a0 + a1 * a1);
  float mean = s * (1.f / 64.f);
  float var  = sq * (1.f / 64.f) - mean * mean;
  float rstd = rsqrtf(var + 1e-5f);
  __bf16* ap = act + (size_t)t * CIN + 256;
  ap[lane]      = (__bf16)((a0 - mean) * rstd * g2[lane] + b2[lane]);
  ap[lane + 32] = (__bf16)((a1 - mean) * rstd * g2[lane + 32] + b2[lane + 32]);
}

// ---------------- token2map scatter (pos-major, coalesced atomics) ----------------
__global__ void scatter_kernel(const float* __restrict__ x, const float* __restrict__ loc,
                               float* __restrict__ buf0) {
  int wave = threadIdx.x >> 5, lane = threadIdx.x & 31;
  int t = blockIdx.x * 8 + wave;
  int b = t >> 14;
  float lx = fminf(fmaxf(loc[(size_t)t * 2 + 0], 0.f), 1.f) * 127.f;
  float ly = fminf(fmaxf(loc[(size_t)t * 2 + 1], 0.f), 1.f) * 127.f;
  int ix = (int)rintf(lx), iy = (int)rintf(ly);
  int pos = iy * HM + ix;
  float* base = buf0 + ((size_t)(b * 16384 + pos)) * 257;
  const float* xp = x + (size_t)t * DDIM;
  for (int c = lane; c < 256; c += 32) atomicAdd(base + c, xp[c]);
  if (lane == 0) atomicAdd(base + 256, 1.0f);
}

// ---------------- normalize: feat = sum/(cnt+eps) * (cnt>0) ----------------
__global__ void norm_kernel(float* __restrict__ buf0) {
  int idx = blockIdx.x * 256 + threadIdx.x;       // < 16,777,216
  int bp = idx >> 8, c = idx & 255;
  float cnt = buf0[(size_t)bp * 257 + 256];
  float v   = buf0[(size_t)bp * 257 + c];
  buf0[(size_t)bp * 257 + c] = (cnt > 0.f) ? v / (cnt + 1e-6f) : 0.f;
}

// ---------------- 3x3 gaussian reconstruct -> buf1 (pos-major, 256ch) ----------------
__global__ void recon_kernel(const float* __restrict__ buf0, float* __restrict__ buf1) {
  const float e1 = 0.8824968934f, e2 = 0.7788007855f;
  const float nrm = 1.f + 4.f * e1 + 4.f * e2;
  const float wC = 1.f / nrm, wE = e1 / nrm, wK = e2 / nrm;
  int idx = blockIdx.x * 256 + threadIdx.x;       // < 16,777,216
  int bp = idx >> 8, c = idx & 255;
  int b = bp >> 14, pos = bp & 16383;
  int py = pos >> 7, px = pos & 127;
  float m = (buf0[(size_t)bp * 257 + 256] > 0.f) ? 1.f : 0.f;
  float f = buf0[(size_t)bp * 257 + c];
  float fF = 0.f, fM = 0.f;
#pragma unroll
  for (int dy = -1; dy <= 1; ++dy) {
#pragma unroll
    for (int dx = -1; dx <= 1; ++dx) {
      int yy = py + dy, xx = px + dx;
      if (yy < 0 || yy >= HM || xx < 0 || xx >= HM) continue;
      float w = (dx == 0 && dy == 0) ? wC : ((dx == 0 || dy == 0) ? wE : wK);
      const float* nb = buf0 + ((size_t)(b * 16384 + yy * HM + xx)) * 257;
      fF += w * nb[c];
      fM += w * ((nb[256] > 0.f) ? 1.f : 0.f);
    }
  }
  float fi = (fM > 0.f) ? fF / (fM + 1e-6f) : 0.f;
  buf1[(size_t)bp * 256 + c] = f + (1.f - m) * fi;
}

// ---------------- bilinear sample of fmap at loc_extra -> act[:,0:256] ----------------
__global__ void sample_kernel(const float* __restrict__ buf1, const float* __restrict__ out2,
                              __bf16* __restrict__ act) {
  int wave = threadIdx.x >> 5, lane = threadIdx.x & 31;
  int t = blockIdx.x * 8 + wave;
  int b = t >> 14, n = t & 16383;
  const float* le = out2 + (size_t)(b * 32768 + 16384 + n) * 2;
  float gx = le[0] * 128.f - 0.5f;
  float gy = le[1] * 128.f - 0.5f;
  float x0 = floorf(gx), y0 = floorf(gy);
  float wx = gx - x0, wy = gy - y0;
  int xi = (int)x0, yi = (int)y0;
  bool vx0 = (xi >= 0) & (xi < HM), vx1 = (xi + 1 >= 0) & (xi + 1 < HM);
  bool vy0 = (yi >= 0) & (yi < HM), vy1 = (yi + 1 >= 0) & (yi + 1 < HM);
  const float* fb = buf1 + (size_t)(b * 16384) * 256;
  float w00 = (1.f - wx) * (1.f - wy), w10 = wx * (1.f - wy);
  float w01 = (1.f - wx) * wy,         w11 = wx * wy;
  __bf16* ap = act + (size_t)t * CIN;
  for (int c = lane; c < 256; c += 32) {
    float v = 0.f;
    if (vx0 && vy0) v += w00 * fb[(size_t)(yi * HM + xi) * 256 + c];
    if (vx1 && vy0) v += w10 * fb[(size_t)(yi * HM + xi + 1) * 256 + c];
    if (vx0 && vy1) v += w01 * fb[(size_t)((yi + 1) * HM + xi) * 256 + c];
    if (vx1 && vy1) v += w11 * fb[(size_t)((yi + 1) * HM + xi + 1) * 256 + c];
    ap[c] = (__bf16)v;
  }
}

// ---------------- repack fc weights into WMMA B-fragment order [kt][n][32] bf16 ----------------
__global__ void repack_kernel(const float* __restrict__ w1, const float* __restrict__ w2,
                              __bf16* __restrict__ w1p, __bf16* __restrict__ w2p) {
  int idx = blockIdx.x * 256 + threadIdx.x;
  if (idx < 10 * 1024 * 32) {
    int kt = idx / (1024 * 32), rem = idx % (1024 * 32);
    int n = rem >> 5, kl = rem & 31;
    w1p[idx] = (__bf16)w1[(size_t)(kt * 32 + kl) * 1024 + n];
  } else {
    int i2 = idx - 10 * 1024 * 32;
    if (i2 < 32 * 256 * 32) {
      int kt = i2 / (256 * 32), rem = i2 % (256 * 32);
      int n = rem >> 5, kl = rem & 31;
      w2p[i2] = (__bf16)w2[(size_t)(kt * 32 + kl) * 256 + n];
    }
  }
}

__device__ __forceinline__ v16bf load_afrag(const __bf16* __restrict__ p) {
  v16bf r;
#pragma unroll
  for (int i = 0; i < 8; ++i) { r[i] = p[i]; r[i + 8] = p[i + 8]; }
  return r;
}
__device__ __forceinline__ v16bf load_bfrag(const __bf16* __restrict__ p) {
  v16bf r;
#pragma unroll
  for (int i = 0; i < 16; ++i) r[i] = p[i];
  return r;
}

// ---------------- WMMA GEMM: 4 waves x (32M x 64N), single-buffered, rolled K loop ----------------
// Per K-step each wave: 2 A-frag + 4 B-frag loads (12 x b128, one clause) -> 8 WMMAs.
// ~130 VGPRs -> no spills, ~7 waves/SIMD hide load latency.
// MODE 0: D = fast_gelu(A@B + bias) -> bf16 (h).  MODE 1: D = A@B + bias -> fp32 out1[b, N+tok, :].
template <int KTILES, int LDA, int LDN, int MODE>
__global__ void gemm_kernel(const __bf16* __restrict__ A, const __bf16* __restrict__ Bp,
                            const float* __restrict__ bias, void* __restrict__ Out) {
  int wave = threadIdx.x >> 5, lane = threadIdx.x & 31;
  int m0 = blockIdx.x * 128 + wave * 32;
  int n0 = blockIdx.y * 64;
  int lh = lane >> 4, lm = lane & 15;

  const __bf16* ap0 = A + (size_t)(m0 + lm) * LDA + lh * 16;      // rows m0..m0+15
  const __bf16* ap1 = ap0 + (size_t)16 * LDA;                      // rows m0+16..m0+31
  const __bf16* bp0 = Bp + (size_t)(n0 + lm) * 32 + lh * 16;       // [kt][n][32] packed

  v8f acc[2][4];
#pragma unroll
  for (int g = 0; g < 2; ++g)
#pragma unroll
    for (int j = 0; j < 4; ++j) acc[g][j] = v8f{};

#pragma unroll 1
  for (int kt = 0; kt < KTILES; ++kt) {
    const __bf16* ak = ap0 + kt * 32;
    const __bf16* bk = bp0 + (size_t)kt * LDN * 32;
    v16bf a0 = load_afrag(ak);
    v16bf a1 = load_afrag(ak + (size_t)16 * LDA);
    v16bf b0 = load_bfrag(bk);
    v16bf b1 = load_bfrag(bk + 16 * 32);
    v16bf b2 = load_bfrag(bk + 32 * 32);
    v16bf b3 = load_bfrag(bk + 48 * 32);
    acc[0][0] = __builtin_amdgcn_wmma_f32_16x16x32_bf16(false, a0, false, b0, (short)0, acc[0][0], false, false);
    acc[1][0] = __builtin_amdgcn_wmma_f32_16x16x32_bf16(false, a1, false, b0, (short)0, acc[1][0], false, false);
    acc[0][1] = __builtin_amdgcn_wmma_f32_16x16x32_bf16(false, a0, false, b1, (short)0, acc[0][1], false, false);
    acc[1][1] = __builtin_amdgcn_wmma_f32_16x16x32_bf16(false, a1, false, b1, (short)0, acc[1][1], false, false);
    acc[0][2] = __builtin_amdgcn_wmma_f32_16x16x32_bf16(false, a0, false, b2, (short)0, acc[0][2], false, false);
    acc[1][2] = __builtin_amdgcn_wmma_f32_16x16x32_bf16(false, a1, false, b2, (short)0, acc[1][2], false, false);
    acc[0][3] = __builtin_amdgcn_wmma_f32_16x16x32_bf16(false, a0, false, b3, (short)0, acc[0][3], false, false);
    acc[1][3] = __builtin_amdgcn_wmma_f32_16x16x32_bf16(false, a1, false, b3, (short)0, acc[1][3], false, false);
  }

#pragma unroll
  for (int g = 0; g < 2; ++g) {
#pragma unroll
    for (int j = 0; j < 4; ++j) {
      int n = n0 + j * 16 + lm;
      float bv = bias[n];
#pragma unroll
      for (int r = 0; r < 8; ++r) {
        int m = m0 + g * 16 + lh * 8 + r;
        float v = acc[g][j][r] + bv;
        if (MODE == 0) {
          // fast GELU: x * sigmoid(1.702 x)  (one v_exp_f32 + rcp, no divergent branch)
          v = v / (1.f + __expf(-1.702f * v));
          ((__bf16*)Out)[(size_t)m * LDN + n] = (__bf16)v;
        } else {
          int b = m >> 14, tok = m & 16383;
          ((float*)Out)[((size_t)(b * 32768 + 16384 + tok)) * 256 + n] = v;
        }
      }
    }
  }
}

extern "C" void kernel_launch(void* const* d_in, const int* in_sizes, int n_in,
                              void* d_out, int out_size, void* d_ws, size_t ws_size,
                              hipStream_t stream) {
  const float* x     = (const float*)d_in[0];
  const float* loc   = (const float*)d_in[1];
  const float* src   = (const float*)d_in[2];
  const float* wd    = (const float*)d_in[7];
  const float* bd    = (const float*)d_in[8];
  const float* n1g   = (const float*)d_in[9];
  const float* n1b   = (const float*)d_in[10];
  const float* convw = (const float*)d_in[11];
  const float* convb = (const float*)d_in[12];
  const float* n2g   = (const float*)d_in[13];
  const float* n2b   = (const float*)d_in[14];
  const float* fc1w  = (const float*)d_in[15];
  const float* fc1b  = (const float*)d_in[16];
  const float* fc2w  = (const float*)d_in[17];
  const float* fc2b  = (const float*)d_in[18];

  float* out1 = (float*)d_out;
  float* out2 = out1 + (size_t)33554432;          // (B, 2N, 256) then (B, 2N, 2)

  char* ws = (char*)d_ws;
  float*  buf0 = (float*)ws;                       //  67,436,544 B : (B,16384,257)
  float*  buf1 = (float*)(ws + 67436544);          //  67,108,864 B : (B,16384,256)
  __bf16* h    = (__bf16*)ws;                      // 134,217,728 B : reuse buf0+buf1 region
  __bf16* act  = (__bf16*)(ws + 134545408);        //  41,943,040 B : (TOK,320)
  __bf16* w1p  = (__bf16*)(ws + 176488448);        //     655,360 B
  __bf16* w2p  = (__bf16*)(ws + 177143808);        //     524,288 B

  zero_kernel<<<65792, 256, 0, stream>>>(buf0, 16842752);
  delta_kernel<<<8192, 256, 0, stream>>>(x, loc, n1g, n1b, wd, bd, out2);
  copyx_kernel<<<16384, 256, 0, stream>>>((const float4*)x, (float4*)out1);
  repack_kernel<<<2304, 256, 0, stream>>>(fc1w, fc2w, w1p, w2p);
  scatter_kernel<<<8192, 256, 0, stream>>>(x, loc, buf0);
  norm_kernel<<<65536, 256, 0, stream>>>(buf0);
  recon_kernel<<<65536, 256, 0, stream>>>(buf0, buf1);
  patch_kernel<<<8192, 256, 0, stream>>>(src, out2, convw, convb, n2g, n2b, act);
  sample_kernel<<<8192, 256, 0, stream>>>(buf1, out2, act);
  // fc1: (65536 x 320) @ (320 x 1024) -> fast_gelu -> h (bf16)
  gemm_kernel<10, CIN, HID, 0><<<dim3(512, 16), 128, 0, stream>>>(act, w1p, fc1b, (void*)h);
  // fc2: (65536 x 1024) @ (1024 x 256) -> out1[b, N+tok, :] (fp32)
  gemm_kernel<32, HID, DDIM, 1><<<dim3(512, 4), 128, 0, stream>>>(h, w2p, fc2b, (void*)out1);
}
